// GraphAttentionLayer_21251498181396
// MI455X (gfx1250) — matrix operand order repaired and checked by reading, not compiled, not run on hardware
//
#include <hip/hip_runtime.h>

typedef __attribute__((ext_vector_type(2)))  float  v2f;
typedef __attribute__((ext_vector_type(8)))  float  v8f;
typedef __attribute__((ext_vector_type(16))) __bf16 v16bf;

#define N_NODES 4096
#define F_DIM   256
#define F_EXT   272   // 256 + 16 pad; col 256 = ones (denominator), 257..271 = 0
#define NEG_SLOPE 0.2f

#define AGG_WAVES   4                    // waves per block in aggregation kernel
#define AGG_THREADS (AGG_WAVES * 32)
#define LDS_ROW_B   80                   // 64B row padded to 80B -> conflict-free
#define BUF_BYTES   (F_EXT * LDS_ROW_B)  // 21760 B per K-strip buffer

// ---------------------------------------------------------------------------
// Kernel 1: Wh = h @ W via f32 WMMA 16x16x4.
//   Writes Wh (f32, row-major, staged in d_out) and WhT (bf16, [F_EXT][N]).
// One wave per 16x16 output tile; 4096 tiles total.
// A layout (32-bit 16x4): lane m=l&15, kb=(l>>4)*2 -> {K=kb, K=kb+1}
// B layout (32-bit 4x16): lane n=l&15, kb=(l>>4)*2 -> {B[kb][n], B[kb+1][n]}
// C layout: lane n=l&15, mb=(l>>4)*8, element r -> C[mb+r][n]
// ---------------------------------------------------------------------------
__global__ __launch_bounds__(256) void gat_gemm1_kernel(
    const float* __restrict__ h, const float* __restrict__ W,
    float* __restrict__ Wh, __bf16* __restrict__ WhT)
{
    int wave = (blockIdx.x * blockDim.x + threadIdx.x) >> 5;
    int lane = threadIdx.x & 31;
    int ct = wave & 15;          // column tile 0..15
    int rt = wave >> 4;          // row strip 0..255
    if (rt >= N_NODES / 16) return;

    int row0 = rt * 16, c0 = ct * 16;
    int m  = lane & 15;
    int n  = lane & 15;
    int kb = (lane >> 4) * 2;

    v8f acc = {};
    const float* hrow = h + (size_t)(row0 + m) * F_DIM;
    #pragma unroll 4
    for (int k = 0; k < F_DIM; k += 4) {
        v2f av, bv;
        av.x = hrow[k + kb];
        av.y = hrow[k + kb + 1];
        const float* wp = W + (size_t)(k + kb) * F_DIM + c0 + n;
        bv.x = wp[0];
        bv.y = wp[F_DIM];
        acc = __builtin_amdgcn_wmma_f32_16x16x4_f32(
            false, av, false, bv, (short)0, acc, false, false);
    }

    int mb = (lane >> 4) * 8;
    #pragma unroll
    for (int r = 0; r < 8; ++r) {
        int row = row0 + mb + r;
        float v = acc[r];
        Wh[(size_t)row * F_DIM + c0 + n] = v;
        WhT[(size_t)(c0 + n) * N_NODES + row] = (__bf16)v;
    }
}

// ---------------------------------------------------------------------------
// Kernel 2: fill WhT extension rows: row 256 = 1.0 (denominator column),
// rows 257..271 = 0.
// ---------------------------------------------------------------------------
__global__ void gat_init_ext_kernel(__bf16* __restrict__ WhT)
{
    int idx = blockIdx.x * blockDim.x + threadIdx.x;
    if (idx < 16 * N_NODES) {
        WhT[(size_t)F_DIM * N_NODES + idx] =
            (idx < N_NODES) ? (__bf16)1.0f : (__bf16)0.0f;
    }
}

// ---------------------------------------------------------------------------
// Kernel 3: e_right[i] = Wh[i,:] . a[F:2F]   (f32, exact)
// ---------------------------------------------------------------------------
__global__ void gat_eright_kernel(const float* __restrict__ Wh,
                                  const float* __restrict__ a,
                                  float* __restrict__ e_right)
{
    int i = blockIdx.x * blockDim.x + threadIdx.x;
    if (i < N_NODES) {
        const float* row = Wh + (size_t)i * F_DIM;
        float acc = 0.0f;
        #pragma unroll 8
        for (int c = 0; c < F_DIM; ++c) acc += row[c] * a[F_DIM + c];
        e_right[i] = acc;
    }
}

// ---------------------------------------------------------------------------
// Kernel 4 (single block): M = max(e_right); s[j] = exp(e_right[j] - M)
// e_left cancels in row-softmax, so s fully determines the attention weights.
// ---------------------------------------------------------------------------
__global__ __launch_bounds__(1024) void gat_sprep_kernel(
    const float* __restrict__ e_right, float* __restrict__ s)
{
    __shared__ float red[1024];
    int t = threadIdx.x;
    float m = -1e30f;
    for (int j = t; j < N_NODES; j += 1024) m = fmaxf(m, e_right[j]);
    red[t] = m;
    __syncthreads();
    for (int w = 512; w > 0; w >>= 1) {
        if (t < w) red[t] = fmaxf(red[t], red[t + w]);
        __syncthreads();
    }
    float M = red[0];
    for (int j = t; j < N_NODES; j += 1024) s[j] = expf(e_right[j] - M);
}

// ---------------------------------------------------------------------------
// Kernel 5: fused aggregation. 4 waves/block, one 16-row strip per wave.
//   acc[ct] (16x16 f32) += A_s(16x32 bf16) x WhT_tile(32x16 bf16), K = 4096.
//   A_s[m][k] = adj[row0+m][k] ? s[k] : 0   built in registers (adj streamed
//   once from HBM = the roofline term; ~2.7us at 23.3 TB/s).
//   B K-strip (272 rows x 32 k, bf16) is SHARED by all 4 waves: staged into
//   LDS with global_load_async_to_lds_b128, double-buffered, overlapped with
//   compute, synced with s_wait_asynccnt + one barrier per K-step. LDS rows
//   padded 64->80B so ds_load_b128 across 16 lanes is bank-conflict-free.
//   Col-tile 16 accumulates the softmax denominator (ones column of Wh_ext).
// ---------------------------------------------------------------------------
__global__ __launch_bounds__(AGG_THREADS) void gat_agg_kernel(
    const int* __restrict__ adj, const float* __restrict__ s,
    const __bf16* __restrict__ WhT, float* __restrict__ out)
{
    __shared__ __align__(16) char bbuf[2][BUF_BYTES];
    __shared__ float dsh[AGG_WAVES][16];

    const int t    = threadIdx.x;         // 0..127
    const int wv   = t >> 5;              // wave 0..3
    const int lane = t & 31;
    const int row0 = (blockIdx.x * AGG_WAVES + wv) * 16;
    const int m    = lane & 15;
    const int n    = lane & 15;
    const int kb   = (lane >> 4) * 16;
    const int koff = (lane >> 4) * 16;

    const unsigned lds_base = (unsigned)(uintptr_t)&bbuf[0][0];

    // Issue async fill of one K-strip: 272 rows x 32 bf16 = 1088 b128 items.
    auto stage = [&](int k0, int p) {
        unsigned base = lds_base + (unsigned)(p * BUF_BYTES);
        #pragma unroll
        for (int j = 0; j < 8; ++j) {
            int i = t + AGG_THREADS * j;
            int c = i >> 2, part = i & 3;
            unsigned ldso = base + (unsigned)(c * LDS_ROW_B + part * 16);
            const __bf16* g = WhT + (size_t)c * N_NODES + k0 + part * 8;
            asm volatile("global_load_async_to_lds_b128 %0, %1, off"
                         :: "v"(ldso), "v"(g) : "memory");
        }
        if (t < 1088 - 8 * AGG_THREADS) {           // remaining 64 items
            int i = 8 * AGG_THREADS + t;
            int c = i >> 2, part = i & 3;
            unsigned ldso = base + (unsigned)(c * LDS_ROW_B + part * 16);
            const __bf16* g = WhT + (size_t)c * N_NODES + k0 + part * 8;
            asm volatile("global_load_async_to_lds_b128 %0, %1, off"
                         :: "v"(ldso), "v"(g) : "memory");
        }
    };

    v8f acc[17];
    #pragma unroll
    for (int ct = 0; ct < 17; ++ct) acc[ct] = (v8f){};

    const int* adjrow = adj + (size_t)(row0 + m) * N_NODES;

    stage(0, 0);
    asm volatile("s_wait_asynccnt 0" ::: "memory");
    __syncthreads();

    int p = 0;
    for (int k0 = 0; k0 < N_NODES; k0 += 32) {
        if (k0 + 32 < N_NODES) stage(k0 + 32, p ^ 1);   // overlap next fill

        if (k0 + 256 < N_NODES)
            __builtin_prefetch(adjrow + k0 + 256, 0, 1);
        int4 q0 = *(const int4*)(adjrow + k0 + kb);
        int4 q1 = *(const int4*)(adjrow + k0 + kb + 4);
        int4 q2 = *(const int4*)(adjrow + k0 + kb + 8);
        int4 q3 = *(const int4*)(adjrow + k0 + kb + 12);
        float4 s0 = *(const float4*)(s + k0 + kb);
        float4 s1 = *(const float4*)(s + k0 + kb + 4);
        float4 s2 = *(const float4*)(s + k0 + kb + 8);
        float4 s3 = *(const float4*)(s + k0 + kb + 12);

        v16bf av;
        av[0]  = (__bf16)(q0.x ? s0.x : 0.0f);
        av[1]  = (__bf16)(q0.y ? s0.y : 0.0f);
        av[2]  = (__bf16)(q0.z ? s0.z : 0.0f);
        av[3]  = (__bf16)(q0.w ? s0.w : 0.0f);
        av[4]  = (__bf16)(q1.x ? s1.x : 0.0f);
        av[5]  = (__bf16)(q1.y ? s1.y : 0.0f);
        av[6]  = (__bf16)(q1.z ? s1.z : 0.0f);
        av[7]  = (__bf16)(q1.w ? s1.w : 0.0f);
        av[8]  = (__bf16)(q2.x ? s2.x : 0.0f);
        av[9]  = (__bf16)(q2.y ? s2.y : 0.0f);
        av[10] = (__bf16)(q2.z ? s2.z : 0.0f);
        av[11] = (__bf16)(q2.w ? s2.w : 0.0f);
        av[12] = (__bf16)(q3.x ? s3.x : 0.0f);
        av[13] = (__bf16)(q3.y ? s3.y : 0.0f);
        av[14] = (__bf16)(q3.z ? s3.z : 0.0f);
        av[15] = (__bf16)(q3.w ? s3.w : 0.0f);

        const char* bufp = &bbuf[p][0];
        #pragma unroll
        for (int ct = 0; ct < 17; ++ct) {
            v16bf bv;   // 32B at 16B alignment -> 2x ds_load_b128
            __builtin_memcpy(&bv, bufp + (ct * 16 + n) * LDS_ROW_B + koff * 2, 32);
            acc[ct] = __builtin_amdgcn_wmma_f32_16x16x32_bf16(
                false, av, false, bv, (short)0, acc[ct], false, false);
        }

        asm volatile("s_wait_asynccnt 0" ::: "memory");  // own fills done
        __syncthreads();                                 // all waves' fills + reads done
        p ^= 1;
    }

    // Broadcast the denominator column (n==0 lanes hold it) within each wave.
    int mb = (lane >> 4) * 8;
    if ((lane & 15) == 0) {
        #pragma unroll
        for (int r = 0; r < 8; ++r) dsh[wv][mb + r] = acc[16][r];
    }
    __syncthreads();

    float d[8];
    #pragma unroll
    for (int r = 0; r < 8; ++r) {
        float dv = dsh[wv][mb + r];
        d[r] = (dv != 0.0f) ? (1.0f / dv) : 0.0f;   // all-masked row -> 0
    }

    #pragma unroll
    for (int ct = 0; ct < 16; ++ct) {
        #pragma unroll
        for (int r = 0; r < 8; ++r) {
            float v = acc[ct][r] * d[r];
            v = (v > 0.0f) ? v : NEG_SLOPE * v;
            out[(size_t)(row0 + mb + r) * F_DIM + ct * 16 + n] = v;
        }
    }
}

// ---------------------------------------------------------------------------
// Host launch: inputs = { h[4096,256] f32, adj[4096,4096] i32,
//                         W[256,256] f32, a[512] f32 } ; out = [4096,256] f32.
// Workspace: WhT bf16 [272][4096] | e_right f32[4096] | s f32[4096].
// Wh f32 is staged in d_out (same size) and overwritten by the final kernel.
// ---------------------------------------------------------------------------
extern "C" void kernel_launch(void* const* d_in, const int* in_sizes, int n_in,
                              void* d_out, int out_size, void* d_ws, size_t ws_size,
                              hipStream_t stream) {
    const float* h   = (const float*)d_in[0];
    const int*   adj = (const int*)d_in[1];
    const float* W   = (const float*)d_in[2];
    const float* a   = (const float*)d_in[3];
    float* out = (float*)d_out;

    char* ws = (char*)d_ws;
    __bf16* WhT    = (__bf16*)ws;                                  // 2,228,224 B
    float* e_right = (float*)(ws + (size_t)F_EXT * N_NODES * 2);   // 16 KB
    float* s_buf   = e_right + N_NODES;                            // 16 KB
    float* Wh      = out;  // staged, overwritten by gat_agg_kernel

    // 1) Wh = h @ W (f32 WMMA), + bf16 transposed copy
    gat_gemm1_kernel<<<dim3((N_NODES / 16) * 16 / 8), dim3(256), 0, stream>>>(
        h, W, Wh, WhT);
    // 2) ones/zero extension rows of WhT
    gat_init_ext_kernel<<<dim3((16 * N_NODES + 255) / 256), dim3(256), 0, stream>>>(WhT);
    // 3) e_right = Wh @ a[F:2F]
    gat_eright_kernel<<<dim3(N_NODES / 256), dim3(256), 0, stream>>>(Wh, a, e_right);
    // 4) s = exp(e_right - max)
    gat_sprep_kernel<<<dim3(1), dim3(1024), 0, stream>>>(e_right, s_buf);
    // 5) fused masked-softmax aggregation (bf16 WMMA, async-LDS B staging)
    gat_agg_kernel<<<dim3(N_NODES / (16 * AGG_WAVES)), dim3(AGG_THREADS), 0, stream>>>(
        adj, s_buf, WhT, out);
}
